// MoEFFN_77214922047954
// MI455X (gfx1250) — matrix operand order repaired and checked by reading, not compiled, not run on hardware
//
#include <hip/hip_runtime.h>
#include <math.h>

// ---------------------------------------------------------------------------
// MoE FFN (top-2) for gfx1250: bf16 WMMA 16x16x32, f32 accumulate.
// B=2, T=2048 -> N=4096 tokens, D=1024, E=8, F=2048.
// ---------------------------------------------------------------------------

typedef __bf16 bf16_t;
typedef __bf16 bf16x16 __attribute__((ext_vector_type(16)));
typedef float  f32x8   __attribute__((ext_vector_type(8)));
typedef int    v4i     __attribute__((ext_vector_type(4)));

#define N_TOK   4096
#define DIM     1024
#define NEXP    8
#define FDIM    2048
#define CAP     N_TOK          // max entries per expert list (token appears once per expert)
#define MT      32             // tokens per tile
#define TILES_MAX (CAP / MT)   // 128
#define XSTR    (DIM  + 8)     // 1032 bf16 row stride -> min-conflict 16B LDS fragment loads
#define HSTR    (FDIM + 8)     // 2056 bf16 row stride

// workspace layout (bytes)
#define OFF_COUNTS  0u                            // int[8]
#define OFF_AUX     32u                           // float cnt1[8], psum[8], sumsq[1]
#define OFF_ETOK    256u                          // int[NEXP*CAP]
#define OFF_EW      (OFF_ETOK + NEXP*CAP*4u)      // float[NEXP*CAP]
#define OFF_XB      (OFF_EW   + NEXP*CAP*4u)      // bf16[N_TOK*DIM]
#define OFF_W1T     (OFF_XB   + (size_t)N_TOK*DIM*2u)          // bf16[NEXP][FDIM][DIM]  (N-major)
#define OFF_W2T     (OFF_W1T  + (size_t)NEXP*FDIM*DIM*2u)      // bf16[NEXP][DIM][FDIM]  (N-major)
#define WS_REQUIRED (OFF_W2T  + (size_t)NEXP*DIM*FDIM*2u)

#define SMEM_BYTES  ((size_t)MT*XSTR*2u + (size_t)MT*HSTR*2u + MT*4u + MT*4u)  // 197888

// --- optional gfx1250 async global->LDS path (guarded; falls back cleanly) ---
#if defined(__has_builtin)
#if __has_builtin(__builtin_amdgcn_global_load_async_to_lds_b128)
#define HAVE_ASYNC_LDS 1
#endif
#if __has_builtin(__builtin_amdgcn_s_wait_asynccnt)
#define HAVE_WAIT_ASYNC_BUILTIN 1
#endif
#endif

#if defined(HAVE_ASYNC_LDS)
typedef __attribute__((address_space(1))) v4i gv4i_t;   // global (AS1) int4
typedef __attribute__((address_space(3))) v4i lv4i_t;   // LDS    (AS3) int4
#endif

__device__ __forceinline__ void wait_asynccnt0() {
#if defined(HAVE_WAIT_ASYNC_BUILTIN)
  __builtin_amdgcn_s_wait_asynccnt(0);
#else
  asm volatile("s_wait_asynccnt 0x0" ::: "memory");
#endif
}

// ---------------------------------------------------------------------------
// WMMA helpers.
// A (16x32 bf16) per-lane layout: lane = {half = lane>>4, m = lane&15};
//   vector elems 0..7  -> K = k0 + half*8 + e
//   vector elems 8..15 -> K = k0 + 16 + half*8 + (e-8)      (two 16B chunks)
// B (32x16 bf16) per-lane layout: lane = {half, n = lane&15};
//   vector elems 0..7 -> K = k0 + half*8 + e; 8..15 -> K = k0 + 16 + half*8 + (e-8)
// C/D (16x16 f32): lane = {half, n = lane&15}; vgpr r -> row m = half*8 + r.
// ---------------------------------------------------------------------------
__device__ __forceinline__ f32x8 wmma_bf16(bf16x16 a, bf16x16 b, f32x8 c) {
  return __builtin_amdgcn_wmma_f32_16x16x32_bf16(false, a, false, b, (short)0, c, false, false);
}

__device__ __forceinline__ bf16x16 load_a_frag(const bf16_t* row, int k0, int kh) {
  bf16x16 a;
  *(uint4*)&a         = *(const uint4*)(row + k0 + kh * 8);
  *(((uint4*)&a) + 1) = *(const uint4*)(row + k0 + 16 + kh * 8);
  return a;
}

__device__ __forceinline__ bf16x16 load_b_frag(const bf16_t* col, int k0, int kh) {
  bf16x16 b;
  *(uint4*)&b         = *(const uint4*)(col + k0 + kh * 8);
  *(((uint4*)&b) + 1) = *(const uint4*)(col + k0 + 16 + kh * 8);
  return b;
}

__device__ __forceinline__ float gelu_erf(float v) {
  return 0.5f * v * (1.0f + erff(v * 0.70710678118654752f));
}

// ---------------------------------------------------------------------------
// Utility kernels
// ---------------------------------------------------------------------------
__global__ void zero_f32_kernel(float* __restrict__ p, size_t n) {
  size_t i = (size_t)blockIdx.x * blockDim.x + threadIdx.x;
  size_t stride = (size_t)gridDim.x * blockDim.x;
  for (; i < n; i += stride) p[i] = 0.0f;
}

__global__ void zero_header_kernel(int* __restrict__ counts, float* __restrict__ auxbuf) {
  int i = threadIdx.x;
  if (i < NEXP) counts[i] = 0;
  if (i < 24) auxbuf[i] = 0.0f;
}

__global__ void convert_x_kernel(const float* __restrict__ x, bf16_t* __restrict__ xb, int n4) {
  int i = blockIdx.x * blockDim.x + threadIdx.x;
  if (i >= n4) return;
  float4 f = ((const float4*)x)[i];
  ushort4 s;
  bf16_t h;
  h = (bf16_t)f.x; s.x = __builtin_bit_cast(unsigned short, h);
  h = (bf16_t)f.y; s.y = __builtin_bit_cast(unsigned short, h);
  h = (bf16_t)f.z; s.z = __builtin_bit_cast(unsigned short, h);
  h = (bf16_t)f.w; s.w = __builtin_bit_cast(unsigned short, h);
  ((ushort4*)xb)[i] = s;
}

// [e][K][N] f32  ->  [e][N][K] bf16   (K, N multiples of 32)
__global__ void transpose_bf16_kernel(const float* __restrict__ in, bf16_t* __restrict__ outp,
                                      int K, int N) {
  __shared__ bf16_t tile[32][33];
  int e = blockIdx.z;
  const float* src = in + (size_t)e * K * N;
  bf16_t* dst = outp + (size_t)e * N * K;
  int n = blockIdx.x * 32 + threadIdx.x;
#pragma unroll
  for (int i = 0; i < 4; ++i) {
    int k = blockIdx.y * 32 + threadIdx.y + i * 8;
    tile[threadIdx.y + i * 8][threadIdx.x] = (bf16_t)src[(size_t)k * N + n];
  }
  __syncthreads();
  int k = blockIdx.y * 32 + threadIdx.x;
#pragma unroll
  for (int i = 0; i < 4; ++i) {
    int nn = blockIdx.x * 32 + threadIdx.y + i * 8;
    dst[(size_t)nn * K + k] = tile[threadIdx.x][threadIdx.y + i * 8];
  }
}

// ---------------------------------------------------------------------------
// Router: one wave per token. logits[8] = x[t,:] @ gate_w; top-2 softmax;
// builds per-expert (token, weight) lists; accumulates aux-loss partials.
// ---------------------------------------------------------------------------
__global__ __launch_bounds__(256)
void router_kernel(const float* __restrict__ x, const float* __restrict__ gw,
                   int* __restrict__ counts, float* __restrict__ cnt1,
                   float* __restrict__ psum, float* __restrict__ sumsq,
                   int* __restrict__ etok, float* __restrict__ ew) {
  int lane = threadIdx.x & 31;
  int wv   = threadIdx.x >> 5;
  int t    = blockIdx.x * 8 + wv;

  float acc[NEXP];
#pragma unroll
  for (int e = 0; e < NEXP; ++e) acc[e] = 0.0f;

  const float* xr = x + (size_t)t * DIM;
  for (int d = lane; d < DIM; d += 32) {
    float xv = xr[d];
    const float4* g = (const float4*)(gw + (size_t)d * NEXP);
    float4 g0 = g[0], g1 = g[1];
    acc[0] = fmaf(xv, g0.x, acc[0]); acc[1] = fmaf(xv, g0.y, acc[1]);
    acc[2] = fmaf(xv, g0.z, acc[2]); acc[3] = fmaf(xv, g0.w, acc[3]);
    acc[4] = fmaf(xv, g1.x, acc[4]); acc[5] = fmaf(xv, g1.y, acc[5]);
    acc[6] = fmaf(xv, g1.z, acc[6]); acc[7] = fmaf(xv, g1.w, acc[7]);
  }
#pragma unroll
  for (int e = 0; e < NEXP; ++e) {
#pragma unroll
    for (int off = 16; off > 0; off >>= 1) acc[e] += __shfl_xor(acc[e], off, 32);
  }

  if (lane == 0) {
    int i0 = 0; float v0 = acc[0];
#pragma unroll
    for (int e = 1; e < NEXP; ++e) if (acc[e] > v0) { v0 = acc[e]; i0 = e; }
    int i1 = (i0 == 0) ? 1 : 0; float v1 = acc[i1];
#pragma unroll
    for (int e = 0; e < NEXP; ++e) if (e != i0 && acc[e] > v1) { v1 = acc[e]; i1 = e; }

    float w0 = 1.0f / (1.0f + expf(v1 - v0));     // softmax over {v0, v1}
    float w1 = 1.0f - w0;

    // full softmax + sum of squares for aux loss
    float s = 0.0f, sq = 0.0f;
    float pr[NEXP];
#pragma unroll
    for (int e = 0; e < NEXP; ++e) { pr[e] = expf(acc[e] - v0); s += pr[e]; sq += acc[e] * acc[e]; }
    float inv = 1.0f / s;
#pragma unroll
    for (int e = 0; e < NEXP; ++e) atomicAdd(&psum[e], pr[e] * inv);
    atomicAdd(sumsq, sq);
    atomicAdd(&cnt1[i0], 1.0f);

    int p0 = atomicAdd(&counts[i0], 1);
    etok[i0 * CAP + p0] = t;  ew[i0 * CAP + p0] = w0;
    int p1 = atomicAdd(&counts[i1], 1);
    etok[i1 * CAP + p1] = t;  ew[i1 * CAP + p1] = w1;
  }
}

__global__ void finalize_aux_kernel(const float* __restrict__ cnt1, const float* __restrict__ psum,
                                    const float* __restrict__ sumsq, float* __restrict__ out_aux) {
  if (threadIdx.x == 0 && blockIdx.x == 0) {
    float a = 0.0f;
    for (int e = 0; e < NEXP; ++e)
      a += (cnt1[e] / (float)N_TOK) * (psum[e] / (float)N_TOK);
    a *= (float)NEXP;
    a += (sumsq[0] / (float)(N_TOK * NEXP)) * 1.0e-3f;
    *out_aux = a;
  }
}

// ---------------------------------------------------------------------------
// Fused expert FFN: block = (expert e, tile of MT=32 tokens from e's list).
// Phase A: H[32,2048] = gelu(X[32,1024] @ W1 + b1)   (H in LDS as bf16)
// Phase B: out += w_tok * (H @ W2 + b2)              (f32 atomics)
// 8 waves; each wave owns a 16-wide N stripe group; 2 M-subtiles.
// ---------------------------------------------------------------------------
__global__ __launch_bounds__(256, 1)
void moe_ffn_kernel(const bf16_t* __restrict__ xb,
                    const bf16_t* __restrict__ w1t, const bf16_t* __restrict__ w2t,
                    const float* __restrict__ b1, const float* __restrict__ b2,
                    const int* __restrict__ counts,
                    const int* __restrict__ etok, const float* __restrict__ ew,
                    float* __restrict__ out) {
  extern __shared__ char smem[];
  bf16_t* Xl  = (bf16_t*)smem;                 // [MT][XSTR]
  bf16_t* Hl  = Xl + MT * XSTR;                // [MT][HSTR]
  float*  wrow = (float*)(Hl + MT * HSTR);     // [MT]
  int*    trow = (int*)(wrow + MT);            // [MT]

  int e    = blockIdx.x >> 7;                  // / TILES_MAX(128)
  int tile = blockIdx.x & (TILES_MAX - 1);
  int cnt  = counts[e];
  int start = tile * MT;
  if (start >= cnt) return;                    // block-uniform early exit

  int tid = threadIdx.x;
  if (tid < MT) {
    int idx = start + tid;
    if (idx < cnt) { trow[tid] = etok[e * CAP + idx]; wrow[tid] = ew[e * CAP + idx]; }
    else           { trow[tid] = 0;                   wrow[tid] = 0.0f; }
  }
  __syncthreads();

  // cooperative X-tile load: thread (row r = tid>>3, seg = tid&7) copies 128 bf16 (256B)
  {
    int r = tid >> 3, seg = tid & 7;
    bf16_t* dstp = Xl + r * XSTR + seg * 128;
    if (start + r < cnt) {
      const bf16_t* srcp = xb + (size_t)trow[r] * DIM + seg * 128;
#if defined(HAVE_ASYNC_LDS)
      // gfx1250 async DMA: global -> LDS, no VGPR round trip (ASYNCcnt tracked)
      gv4i_t* g = (gv4i_t*)srcp;
      lv4i_t* l = (lv4i_t*)dstp;
#pragma unroll
      for (int i = 0; i < 16; ++i)
        __builtin_amdgcn_global_load_async_to_lds_b128(g + i, l + i, 0, 0);
#else
      const uint4* src = (const uint4*)srcp;
      uint4* dst = (uint4*)dstp;
#pragma unroll
      for (int i = 0; i < 16; ++i) dst[i] = src[i];
#endif
    } else {
      uint4 z = {0u, 0u, 0u, 0u};
      uint4* dst = (uint4*)dstp;
#pragma unroll
      for (int i = 0; i < 16; ++i) dst[i] = z;
    }
  }
#if defined(HAVE_ASYNC_LDS)
  wait_asynccnt0();
#endif
  __syncthreads();

  int lane = tid & 31, wv = tid >> 5;
  int kh = lane >> 4, lm = lane & 15;          // half-wave, lane-in-half

  // ---------------- Phase A: H = gelu(X @ W1 + b1) ----------------
  const bf16_t* Ar0 = Xl + lm * XSTR;
  const bf16_t* Ar1 = Xl + (16 + lm) * XSTR;
  for (int j = 0; j < FDIM / (16 * 8); ++j) {  // 16 n-subtiles per wave
    int n0 = (wv * 16 + j) * 16;
    const bf16_t* Bc = w1t + ((size_t)e * FDIM + n0 + lm) * (size_t)DIM;
    f32x8 acc0 = {}, acc1 = {};
#pragma unroll 4
    for (int k0 = 0; k0 < DIM; k0 += 32) {
      bf16x16 b  = load_b_frag(Bc, k0, kh);
      bf16x16 a0 = load_a_frag(Ar0, k0, kh);
      bf16x16 a1 = load_a_frag(Ar1, k0, kh);
      acc0 = wmma_bf16(a0, b, acc0);
      acc1 = wmma_bf16(a1, b, acc1);
    }
    float bias = b1[(size_t)e * FDIM + n0 + lm];
#pragma unroll
    for (int r = 0; r < 8; ++r) {
      int m0 = kh * 8 + r;
      float v = acc0[r] + bias;
      Hl[m0 * HSTR + n0 + lm] = (bf16_t)gelu_erf(v);
      v = acc1[r] + bias;
      Hl[(16 + m0) * HSTR + n0 + lm] = (bf16_t)gelu_erf(v);
    }
  }
  __syncthreads();

  // ---------------- Phase B: out += w * (H @ W2 + b2) ----------------
  const bf16_t* Hr0 = Hl + lm * HSTR;
  const bf16_t* Hr1 = Hl + (16 + lm) * HSTR;
  for (int j = 0; j < DIM / (16 * 8); ++j) {   // 8 n-subtiles per wave
    int n0 = (wv * 8 + j) * 16;
    const bf16_t* Bc = w2t + ((size_t)e * DIM + n0 + lm) * (size_t)FDIM;
    f32x8 acc0 = {}, acc1 = {};
#pragma unroll 4
    for (int k0 = 0; k0 < FDIM; k0 += 32) {
      bf16x16 b  = load_b_frag(Bc, k0, kh);
      bf16x16 a0 = load_a_frag(Hr0, k0, kh);
      bf16x16 a1 = load_a_frag(Hr1, k0, kh);
      acc0 = wmma_bf16(a0, b, acc0);
      acc1 = wmma_bf16(a1, b, acc1);
    }
    int col = n0 + lm;
    float bias = b2[(size_t)e * DIM + col];
#pragma unroll
    for (int r = 0; r < 8; ++r) {
      int m0 = kh * 8 + r;
      float w = wrow[m0];
      if (w != 0.0f) atomicAdd(out + (size_t)trow[m0] * DIM + col, w * (acc0[r] + bias));
      int m1 = 16 + m0;
      w = wrow[m1];
      if (w != 0.0f) atomicAdd(out + (size_t)trow[m1] * DIM + col, w * (acc1[r] + bias));
    }
  }
}

// ---------------------------------------------------------------------------
extern "C" void kernel_launch(void* const* d_in, const int* in_sizes, int n_in,
                              void* d_out, int out_size, void* d_ws, size_t ws_size,
                              hipStream_t stream) {
  const float* x  = (const float*)d_in[0];
  const float* gw = (const float*)d_in[1];
  const float* w1 = (const float*)d_in[2];
  const float* b1 = (const float*)d_in[3];
  const float* w2 = (const float*)d_in[4];
  const float* b2 = (const float*)d_in[5];
  float* out = (float*)d_out;

  if (ws_size < WS_REQUIRED) return;   // need ~76 MB scratch

  char* ws = (char*)d_ws;
  int*    counts = (int*)(ws + OFF_COUNTS);
  float*  auxbuf = (float*)(ws + OFF_AUX);
  float*  cnt1   = auxbuf;        // [8]
  float*  psum   = auxbuf + 8;    // [8]
  float*  sumsq  = auxbuf + 16;   // [1]
  int*    etok   = (int*)(ws + OFF_ETOK);
  float*  ew     = (float*)(ws + OFF_EW);
  bf16_t* xb     = (bf16_t*)(ws + OFF_XB);
  bf16_t* w1t    = (bf16_t*)(ws + OFF_W1T);
  bf16_t* w2t    = (bf16_t*)(ws + OFF_W2T);

  zero_header_kernel<<<1, 64, 0, stream>>>(counts, auxbuf);
  zero_f32_kernel<<<2048, 256, 0, stream>>>(out, (size_t)N_TOK * DIM + 1);

  convert_x_kernel<<<(N_TOK * DIM / 4 + 255) / 256, 256, 0, stream>>>(x, xb, N_TOK * DIM / 4);
  dim3 tb(32, 8);
  transpose_bf16_kernel<<<dim3(FDIM / 32, DIM / 32, NEXP), tb, 0, stream>>>(w1, w1t, DIM, FDIM);
  transpose_bf16_kernel<<<dim3(DIM / 32, FDIM / 32, NEXP), tb, 0, stream>>>(w2, w2t, FDIM, DIM);

  router_kernel<<<N_TOK / 8, 256, 0, stream>>>(x, gw, counts, cnt1, psum, sumsq, etok, ew);
  finalize_aux_kernel<<<1, 32, 0, stream>>>(cnt1, psum, sumsq, out + (size_t)N_TOK * DIM);

  (void)hipFuncSetAttribute((const void*)moe_ffn_kernel,
                            hipFuncAttributeMaxDynamicSharedMemorySize, (int)SMEM_BYTES);
  moe_ffn_kernel<<<NEXP * TILES_MAX, 256, SMEM_BYTES, stream>>>(
      xb, w1t, w2t, b1, b2, counts, etok, ew, out);
}
